// MultiDilateLocalAttention_70257075028734
// MI455X (gfx1250) — compile-verified, matrix operand order checked by reading, not run on hardware
//
#include <hip/hip_runtime.h>

// ---------------------------------------------------------------------------
// Multi-dilated local attention for MI455X (gfx1250, wave32, WMMA).
//   B=8, N=3072, C=768, heads=12, dilations=3, ks=3 -> hd=64, d=256, h=4, nb=1024
// Pipeline:
//   1) convert x / qkv_w / proj_w to bf16 in workspace
//   2) qkv GEMM  (24576x768)x(2304x768)^T  -> bf16  (v_wmma_f32_16x16x32_bf16)
//   3) windowed 3x3 attention, softmax in fp32, scatter to (B,N,C) bf16
//   4) proj GEMM (24576x768)x(768x768)^T + bias -> fp32 d_out
// GEMM: wave32 owns a 32x64 tile (2x4 wmma accumulators -> 8 WMMA / 6 frags),
// 8 waves -> 128x128 block tile; WGP-scope prefetch of the K slab two ahead.
// ---------------------------------------------------------------------------

typedef __attribute__((ext_vector_type(16))) __bf16 bf16x16;
typedef __attribute__((ext_vector_type(8)))  __bf16 bf16x8;
typedef __attribute__((ext_vector_type(8)))  float  floatx8;

#define MTOT  24576      // B*N
#define KDIM  768        // C
#define NQKV  2304       // 3*C
#define NPRJ  768        // C
#define SEQ   3072       // N
#define NBW   1024       // nb = N/ks

static __device__ inline unsigned short f2bf_bits(float f) {
  union { float f; unsigned u; } v; v.f = f;
  unsigned r = v.u + 0x7FFFu + ((v.u >> 16) & 1u);   // round-to-nearest-even
  return (unsigned short)(r >> 16);
}
static __device__ inline __bf16 f2bf(float f) {
  unsigned short h = f2bf_bits(f);
  return __builtin_bit_cast(__bf16, h);
}
static __device__ inline float bf2f_bits(unsigned h) {
  union { unsigned u; float f; } v; v.u = h << 16;
  return v.f;
}

// --------------------------- fp32 -> bf16 convert ---------------------------
__global__ void cvt_f32_to_bf16(const float* __restrict__ in,
                                __bf16* __restrict__ out, int n) {
  for (int i = blockIdx.x * blockDim.x + threadIdx.x; i < n;
       i += gridDim.x * blockDim.x)
    out[i] = f2bf(in[i]);
}

// ------------------------- WMMA bf16 fragment load --------------------------
// 16-bit A/B 16x32 layout: lane L (<16) holds row L, K = {khalf..khalf+7} in
// v0..3 and {khalf+16..+23} in v4..7 with khalf = (L>=16)*8.  Both A (rows of
// activations) and B (rows of the weight matrix, since we compute X @ W^T)
// load identically from row-major K-contiguous storage.
static __device__ inline bf16x16 load_frag(const __bf16* __restrict__ p) {
  bf16x8 lo = *reinterpret_cast<const bf16x8*>(p);        // K..K+7
  bf16x8 hi = *reinterpret_cast<const bf16x8*>(p + 16);   // K+16..K+23
  return __builtin_shufflevector(lo, hi, 0, 1, 2, 3, 4, 5, 6, 7,
                                 8, 9, 10, 11, 12, 13, 14, 15);
}

// ------------------------------- WMMA GEMM ---------------------------------
// C(M x Nc) = A(M x K) @ W(Nc x K)^T
// wave -> 32x64 tile (2 M-subtiles x 4 N-subtiles), 8 waves (4M x 2N)
// -> 128x128 block tile.
template <bool BF16_OUT>
__global__ __launch_bounds__(256) void gemm_wmma_bf16(
    const __bf16* __restrict__ A, const __bf16* __restrict__ W, int K, int Nc,
    __bf16* __restrict__ outB, float* __restrict__ outF,
    const float* __restrict__ bias) {
  const int lane  = threadIdx.x & 31;
  const int wave  = threadIdx.x >> 5;
  const int m0    = blockIdx.x * 128 + (wave & 3) * 32;
  const int n0    = blockIdx.y * 128 + (wave >> 2) * 64;
  const int rlo   = lane & 15;
  const int khalf = (lane >> 4) * 8;

  const __bf16* arow[2];
  const __bf16* brow[4];
#pragma unroll
  for (int i = 0; i < 2; ++i)
    arow[i] = A + (size_t)(m0 + 16 * i + rlo) * K + khalf;
#pragma unroll
  for (int j = 0; j < 4; ++j)
    brow[j] = W + (size_t)(n0 + 16 * j + rlo) * K + khalf;

  floatx8 acc[2][4];
#pragma unroll
  for (int i = 0; i < 2; ++i)
#pragma unroll
    for (int j = 0; j < 4; ++j) acc[i][j] = floatx8{};

  for (int k0 = 0; k0 < K; k0 += 32) {
    bf16x16 af[2], bf[4];
#pragma unroll
    for (int i = 0; i < 2; ++i) af[i] = load_frag(arow[i] + k0);
#pragma unroll
    for (int j = 0; j < 4; ++j) bf[j] = load_frag(brow[j] + k0);
    // WGP-scope prefetch of the K slab two ahead (pulls into all cache levels)
    __builtin_prefetch(arow[0] + k0 + 64, 0, 3);
    __builtin_prefetch(arow[1] + k0 + 64, 0, 3);
    __builtin_prefetch(brow[0] + k0 + 64, 0, 3);
    __builtin_prefetch(brow[2] + k0 + 64, 0, 3);
#pragma unroll
    for (int i = 0; i < 2; ++i)
#pragma unroll
      for (int j = 0; j < 4; ++j)
        acc[i][j] = __builtin_amdgcn_wmma_f32_16x16x32_bf16(
            false, af[i], false, bf[j], (short)0, acc[i][j], false, false);
  }

  // C/D layout: lane<16 -> col=lane, rows 0..7 in v0..7 ; lane>=16 -> col=
  // lane-16, rows 8..15.
  const int nofs = lane & 15;
  const int mofs = (lane >> 4) * 8;
#pragma unroll
  for (int i = 0; i < 2; ++i)
#pragma unroll
    for (int j = 0; j < 4; ++j) {
      const int mt = m0 + 16 * i;
      const int nt = n0 + 16 * j;
#pragma unroll
      for (int e = 0; e < 8; ++e) {
        const size_t idx = (size_t)(mt + mofs + e) * Nc + (nt + nofs);
        if constexpr (BF16_OUT)
          outB[idx] = f2bf(acc[i][j][e]);
        else
          outF[idx] = acc[i][j][e] + bias[nt + nofs];
      }
    }
}

// --------------------------- windowed attention -----------------------------
// One wave32 per (dilation nd_i, batch b, head h_i, window nb_i).
// Each lane owns 2 consecutive channels of hd=64; 3x3 score matrix via
// cross-lane reduction; softmax fp32; out = attn @ v; scatter-store matching
// the reference's final reshape directly in (B, N, C) order.
__global__ __launch_bounds__(256) void attn_dil3(
    const __bf16* __restrict__ qkv, __bf16* __restrict__ y) {
  const int lane = threadIdx.x & 31;
  const int wave = threadIdx.x >> 5;
  int Wid = blockIdx.x * 8 + wave;        // [0, 3*8*4*1024)
  const int nb_i = Wid & (NBW - 1);  Wid >>= 10;
  const int h_i  = Wid & 3;          Wid >>= 2;
  const int b    = Wid & 7;
  const int nd_i = Wid >> 3;

  const int cb = nd_i * 256 + h_i * 64 + 2 * lane;  // channel pair base

  float qx[3], qy[3], kx[3], ky[3], vx[3], vy[3];
#pragma unroll
  for (int w = 0; w < 3; ++w) {
    const size_t row = (size_t)(b * SEQ + w * NBW + nb_i) * NQKV;
    const unsigned qp = *reinterpret_cast<const unsigned*>(qkv + row + cb);
    const unsigned kp = *reinterpret_cast<const unsigned*>(qkv + row + 768 + cb);
    const unsigned vp = *reinterpret_cast<const unsigned*>(qkv + row + 1536 + cb);
    qx[w] = bf2f_bits(qp & 0xffffu); qy[w] = bf2f_bits(qp >> 16);
    kx[w] = bf2f_bits(kp & 0xffffu); ky[w] = bf2f_bits(kp >> 16);
    vx[w] = bf2f_bits(vp & 0xffffu); vy[w] = bf2f_bits(vp >> 16);
  }

  float s[9];
#pragma unroll
  for (int w = 0; w < 3; ++w)
#pragma unroll
    for (int u = 0; u < 3; ++u)
      s[w * 3 + u] = qx[w] * kx[u] + qy[w] * ky[u];

#pragma unroll
  for (int off = 16; off > 0; off >>= 1)
#pragma unroll
    for (int i = 0; i < 9; ++i) s[i] += __shfl_xor(s[i], off, 32);

  float a[9];
#pragma unroll
  for (int w = 0; w < 3; ++w) {
    const float s0 = s[w * 3 + 0] * 0.125f;   // hd^-0.5 = 1/8
    const float s1 = s[w * 3 + 1] * 0.125f;
    const float s2 = s[w * 3 + 2] * 0.125f;
    const float m  = fmaxf(s0, fmaxf(s1, s2));
    const float e0 = __expf(s0 - m), e1 = __expf(s1 - m), e2 = __expf(s2 - m);
    const float r  = 1.0f / (e0 + e1 + e2);
    a[w * 3 + 0] = e0 * r; a[w * 3 + 1] = e1 * r; a[w * 3 + 2] = e2 * r;
  }

#pragma unroll
  for (int w = 0; w < 3; ++w) {
    const float ox = a[w*3+0]*vx[0] + a[w*3+1]*vx[1] + a[w*3+2]*vx[2];
    const float oy = a[w*3+0]*vy[0] + a[w*3+1]*vy[1] + a[w*3+2]*vy[2];
    // reference's flatten of (nb,h,ks,hd) -> (N,d), then channel nd_i*256+d
    const int fl  = h_i * 192 + w * 64 + 2 * lane;
    const int N_o = nb_i * 3 + (fl >> 8);
    const int col = nd_i * 256 + (fl & 255);
    const unsigned pk = (unsigned)f2bf_bits(ox) |
                        ((unsigned)f2bf_bits(oy) << 16);
    *reinterpret_cast<unsigned*>(y + (size_t)(b * SEQ + N_o) * KDIM + col) = pk;
  }
}

// ------------------------------- launcher -----------------------------------
extern "C" void kernel_launch(void* const* d_in, const int* in_sizes, int n_in,
                              void* d_out, int out_size, void* d_ws,
                              size_t ws_size, hipStream_t stream) {
  (void)in_sizes; (void)n_in; (void)out_size; (void)ws_size;
  const float* x      = (const float*)d_in[0];
  const float* qkv_w  = (const float*)d_in[1];
  const float* proj_w = (const float*)d_in[2];
  const float* proj_b = (const float*)d_in[3];
  float* out = (float*)d_out;

  __bf16* xb     = (__bf16*)d_ws;                       // 24576*768
  __bf16* wqkvb  = xb    + (size_t)MTOT * KDIM;         // 2304*768
  __bf16* wprojb = wqkvb + (size_t)NQKV * KDIM;         // 768*768
  __bf16* qkvb   = wprojb + (size_t)NPRJ * KDIM;        // 24576*2304
  __bf16* yb     = qkvb  + (size_t)MTOT * NQKV;         // 24576*768

  cvt_f32_to_bf16<<<4096, 256, 0, stream>>>(x, xb, MTOT * KDIM);
  cvt_f32_to_bf16<<<1024, 256, 0, stream>>>(qkv_w, wqkvb, NQKV * KDIM);
  cvt_f32_to_bf16<<<512, 256, 0, stream>>>(proj_w, wprojb, NPRJ * KDIM);

  gemm_wmma_bf16<true><<<dim3(MTOT / 128, NQKV / 128), 256, 0, stream>>>(
      xb, wqkvb, KDIM, NQKV, qkvb, nullptr, nullptr);

  attn_dil3<<<(3 * 8 * 4 * NBW) / 8, 256, 0, stream>>>(qkvb, yb);

  gemm_wmma_bf16<false><<<dim3(MTOT / 128, NPRJ / 128), 256, 0, stream>>>(
      yb, wprojb, KDIM, NPRJ, nullptr, out, proj_b);
}
